// ReinforcemerRepacker_67370857005364
// MI455X (gfx1250) — matrix operand order repaired and checked by reading, not compiled, not run on hardware
//
#include <hip/hip_runtime.h>
#include <hip/hip_bf16.h>

// ---------------------------------------------------------------------------
// Problem constants (from reference)
// ---------------------------------------------------------------------------
#define N_NODES 10000
#define E_EDGES 300000
#define DD      128
#define DIN_    384
#define NH      4
#define NL      3
#define NBINS_  16
#define BMAX_   20.0f

typedef __bf16 bf16_t;
typedef bf16_t v16bf __attribute__((ext_vector_type(16)));
typedef bf16_t v8bf  __attribute__((ext_vector_type(8)));
typedef float  v8f   __attribute__((ext_vector_type(8)));

// ---------------------------------------------------------------------------
// Helpers
// ---------------------------------------------------------------------------
__device__ __forceinline__ bf16_t tobf(float x) {
  unsigned u = __builtin_bit_cast(unsigned, x);
  unsigned r = (u + 0x7FFFu + ((u >> 16) & 1u)) >> 16;   // RNE
  return __builtin_bit_cast(bf16_t, (unsigned short)r);
}

__device__ __forceinline__ float gelu_f(float x) {
  return 0.5f * x * (1.0f + erff(x * 0.70710678118654752f));
}

// order-preserving float <-> uint for atomic segment-max
__device__ __forceinline__ unsigned fenc(float f) {
  unsigned u = __builtin_bit_cast(unsigned, f);
  return (u & 0x80000000u) ? ~u : (u | 0x80000000u);
}
__device__ __forceinline__ float fdec(unsigned u) {
  unsigned v = (u & 0x80000000u) ? (u & 0x7FFFFFFFu) : ~u;
  return __builtin_bit_cast(float, v);
}

__device__ __forceinline__ v8f wmma_bf16(v16bf a, v16bf b, v8f c) {
  return __builtin_amdgcn_wmma_f32_16x16x32_bf16(
      /*neg_a=*/false, a, /*neg_b=*/false, b,
      /*c_mod=*/(short)0, c, /*reuse_a=*/false, /*reuse_b=*/false);
}

// A-fragment (16x32 bf16) from a row-major LDS tile [16][strideH halves].
// ISA 16-bit A layout: lanes 0-15 hold K=0..7 / 16..23, lanes 16-31 hold
// K=8..15 / 24..31 of the 32-wide K slice. Two 16B LDS loads per lane.
__device__ __forceinline__ v16bf load_a_frag(const bf16_t* tile, int strideH,
                                             int kstep, int lane) {
  int m  = lane & 15;
  int hi = lane >> 4;
  const bf16_t* p = tile + m * strideH + kstep * 32 + hi * 8;
  v8bf lo = *(const v8bf*)p;
  v8bf hv = *(const v8bf*)(p + 16);
  v16bf r;
#pragma unroll
  for (int i = 0; i < 8; ++i) { r[i] = lo[i]; r[i + 8] = hv[i]; }
  return r;
}

// B-fragment (32x16 bf16) from pre-transposed global weights Wt[n][k]
// (row n = output column, strideH halves). Lane n = nbase + (lane&15),
// lanes 0-15 hold K=0..7 / 16..23, lanes 16-31 hold K=8..15 / 24..31.
__device__ __forceinline__ v16bf load_b_frag(const bf16_t* wt, int strideH,
                                             int nbase, int kstep, int lane) {
  const bf16_t* p = wt + (size_t)(nbase + (lane & 15)) * strideH +
                    kstep * 32 + (lane >> 4) * 8;
  v8bf lo = *(const v8bf*)p;
  v8bf hv = *(const v8bf*)(p + 16);
  v16bf r;
#pragma unroll
  for (int i = 0; i < 8; ++i) { r[i] = lo[i]; r[i + 8] = hv[i]; }
  return r;
}

// Gather feats tile [16 edges][384] bf16 into LDS (stride 392 halves),
// 8 halves (16B) per global load, fully cooperative across 128 threads.
__device__ __forceinline__ void gather_feats(bf16_t* sA, int eb, int tid,
                                             const int* src, const int* snk,
                                             const bf16_t* bbh,
                                             const bf16_t* eah) {
  for (int i = tid; i < 16 * 48; i += 128) {
    int r = i / 48, c8 = (i % 48) * 8;
    int e = eb + r;
    const bf16_t* p;
    if (c8 < 128)      p = bbh + (size_t)src[e] * DD + c8;
    else if (c8 < 256) p = bbh + (size_t)snk[e] * DD + (c8 - 128);
    else               p = eah + (size_t)e * DD + (c8 - 256);
    *(v8bf*)(sA + r * 392 + c8) = *(const v8bf*)p;
  }
}

// ---------------------------------------------------------------------------
// Small prep kernels
// ---------------------------------------------------------------------------
__global__ void k_copy_x(const float* __restrict__ x, float* bbf, bf16_t* bbh) {
  int i = blockIdx.x * 256 + threadIdx.x;
  if (i < N_NODES * DD) { float v = x[i]; bbf[i] = v; bbh[i] = tobf(v); }
}

// RBF encode + edge embedding: eattr = exp(-((d-mu)/sigma)^2)+1e-8 @ W + b
__global__ __launch_bounds__(128) void k_rbf(const float* __restrict__ dist,
                                             const float* __restrict__ ew,
                                             const float* __restrict__ eb,
                                             float* eaf, bf16_t* eah) {
  __shared__ float sr[8 * 16];
  __shared__ float sw[16 * DD];
  int tid = threadIdx.x;
  int e0 = blockIdx.x * 8;
  for (int i = tid; i < 16 * DD; i += 128) sw[i] = ew[i];
  {
    int el = tid >> 4, b = tid & 15;
    float mu = (BMAX_ / (NBINS_ - 1)) * b;   // linspace(0, 20, 16)
    float sigma = BMAX_ / NBINS_;            // 1.25
    float t = (dist[e0 + el] - mu) / sigma;
    sr[tid] = expf(-t * t) + 1e-8f;
  }
  __syncthreads();
  for (int i = tid; i < 8 * DD; i += 128) {
    int el = i >> 7, d = i & 127;
    float acc = eb[d];
#pragma unroll
    for (int b = 0; b < 16; ++b) acc += sr[el * 16 + b] * sw[b * DD + d];
    size_t o = (size_t)(e0 + el) * DD + d;
    eaf[o] = acc; eah[o] = tobf(acc);
  }
}

// Transpose fp32 [K][Nd] -> bf16 [Nd][K]
__global__ void kT(bf16_t* dst, const float* __restrict__ src, int K, int Nd) {
  int i = blockIdx.x * 256 + threadIdx.x;
  if (i < K * Nd) {
    int k = i / Nd, n = i % Nd;
    dst[(size_t)n * K + k] = tobf(src[i]);
  }
}

__global__ void k_init(unsigned* segm, float* segn, float* upd) {
  int i = blockIdx.x * 256 + threadIdx.x;
  if (i < N_NODES * NH) { segm[i] = fenc(-3.0e38f); segn[i] = 0.0f; }
  if (i < N_NODES * DD) upd[i] = 0.0f;
}

// ---------------------------------------------------------------------------
// Fused edge kernel: gather -> [attention GEMM (K=384,N=512) + n1 GEMM
// (K=384,N=128)] -> leakyrelu+att-dot / gelu -> n2 GEMM -> gelu -> n3 GEMM.
// One 16-edge tile per 128-thread (4-wave) block; each wave owns 1 head
// (8 C tiles) + 2 n1 tiles in a single K loop for A-fragment reuse.
// ---------------------------------------------------------------------------
__global__ __launch_bounds__(128) void k_edge_stage1(
    const int* __restrict__ src, const int* __restrict__ snk,
    const bf16_t* __restrict__ bbh, const bf16_t* __restrict__ eah,
    const bf16_t* __restrict__ W1t, const bf16_t* __restrict__ n2t,
    const bf16_t* __restrict__ n3t,
    const float* __restrict__ aWb, const float* __restrict__ aAg,
    const float* __restrict__ aAb,
    const float* __restrict__ n1b, const float* __restrict__ n2b,
    const float* __restrict__ n3b,
    float* watt, float* nupd) {
  __shared__ __attribute__((aligned(16))) bf16_t sA[16 * 392];
  __shared__ __attribute__((aligned(16))) bf16_t sM[16 * 136];
  __shared__ __attribute__((aligned(16))) bf16_t sM2[16 * 136];
  __shared__ float sAA[NH * DD];

  int tid = threadIdx.x, lane = tid & 31, w = tid >> 5;
  int eb = blockIdx.x * 16;

  __builtin_prefetch(W1t + (size_t)w * 160 * DIN_, 0, 1);  // warm weight slice

  gather_feats(sA, eb, tid, src, snk, bbh, eah);
  for (int i = tid; i < NH * DD; i += 128) sAA[i] = aAg[i];
  __syncthreads();

  // ---- fused GEMM: 10 output tiles per wave ----
  v8f c[10];
#pragma unroll
  for (int t = 0; t < 10; ++t)
#pragma unroll
    for (int i = 0; i < 8; ++i) c[t][i] = 0.0f;

  int nb[10];
#pragma unroll
  for (int t = 0; t < 8; ++t) nb[t] = w * 128 + t * 16;   // attention head w
  nb[8] = 512 + w * 32;                                    // n1 cols
  nb[9] = 512 + w * 32 + 16;

  for (int ks = 0; ks < 12; ++ks) {
    v16bf a = load_a_frag(sA, 392, ks, lane);
#pragma unroll
    for (int t = 0; t < 10; ++t) {
      v16bf b = load_b_frag(W1t, DIN_, nb[t], ks, lane);
      c[t] = wmma_bf16(a, b, c[t]);
    }
  }

  int rbase = (lane >> 4) * 8;

  // ---- attention epilogue: bias + leakyrelu(0.2) + dot with aA, reduce ----
  float acc[8];
#pragma unroll
  for (int v = 0; v < 8; ++v) acc[v] = 0.0f;
#pragma unroll
  for (int t = 0; t < 8; ++t) {
    int col = w * 128 + t * 16 + (lane & 15);
    float bias = aWb[col];
    float aa = sAA[col];
#pragma unroll
    for (int v = 0; v < 8; ++v) {
      float x = c[t][v] + bias;
      x = (x > 0.0f) ? x : 0.2f * x;
      acc[v] += x * aa;
    }
  }
#pragma unroll
  for (int v = 0; v < 8; ++v)
    for (int m = 1; m < 16; m <<= 1) acc[v] += __shfl_xor(acc[v], m, 32);
  if ((lane & 15) == 0) {
    float ab = aAb[w];
#pragma unroll
    for (int v = 0; v < 8; ++v)
      watt[(size_t)(eb + rbase + v) * NH + w] = acc[v] + ab;
  }

  // ---- n1 epilogue: bias + gelu -> LDS bf16 ----
#pragma unroll
  for (int t = 8; t < 10; ++t) {
    int col = w * 32 + (t - 8) * 16 + (lane & 15);
    float bias = n1b[col];
#pragma unroll
    for (int v = 0; v < 8; ++v)
      sM[(rbase + v) * 136 + col] = tobf(gelu_f(c[t][v] + bias));
  }
  __syncthreads();

  // ---- n2: [16,128] @ [128,128], gelu ----
  v8f c2[2];
#pragma unroll
  for (int t = 0; t < 2; ++t)
#pragma unroll
    for (int i = 0; i < 8; ++i) c2[t][i] = 0.0f;
  for (int ks = 0; ks < 4; ++ks) {
    v16bf a = load_a_frag(sM, 136, ks, lane);
#pragma unroll
    for (int t = 0; t < 2; ++t) {
      v16bf b = load_b_frag(n2t, DD, w * 32 + t * 16, ks, lane);
      c2[t] = wmma_bf16(a, b, c2[t]);
    }
  }
#pragma unroll
  for (int t = 0; t < 2; ++t) {
    int col = w * 32 + t * 16 + (lane & 15);
    float bias = n2b[col];
#pragma unroll
    for (int v = 0; v < 8; ++v)
      sM2[(rbase + v) * 136 + col] = tobf(gelu_f(c2[t][v] + bias));
  }
  __syncthreads();

  // ---- n3: [16,128] @ [128,128] + bias -> nupd (fp32, global) ----
#pragma unroll
  for (int t = 0; t < 2; ++t)
#pragma unroll
    for (int i = 0; i < 8; ++i) c2[t][i] = 0.0f;
  for (int ks = 0; ks < 4; ++ks) {
    v16bf a = load_a_frag(sM2, 136, ks, lane);
#pragma unroll
    for (int t = 0; t < 2; ++t) {
      v16bf b = load_b_frag(n3t, DD, w * 32 + t * 16, ks, lane);
      c2[t] = wmma_bf16(a, b, c2[t]);
    }
  }
#pragma unroll
  for (int t = 0; t < 2; ++t) {
    int col = w * 32 + t * 16 + (lane & 15);
    float bias = n3b[col];
#pragma unroll
    for (int v = 0; v < 8; ++v)
      nupd[(size_t)(eb + rbase + v) * DD + col] = c2[t][v] + bias;
  }
}

// ---------------------------------------------------------------------------
// Scatter softmax
// ---------------------------------------------------------------------------
__global__ void k_seg_max(const int* __restrict__ snk,
                          const float* __restrict__ watt, unsigned* segm) {
  int i = blockIdx.x * 256 + threadIdx.x;
  if (i < E_EDGES * NH) {
    int e = i >> 2, h = i & 3;
    atomicMax(&segm[(size_t)snk[e] * NH + h], fenc(watt[i]));
  }
}

__global__ void k_att_exp(const int* __restrict__ snk, float* watt,
                          const unsigned* __restrict__ segm, float* segn) {
  int i = blockIdx.x * 256 + threadIdx.x;
  if (i < E_EDGES * NH) {
    int e = i >> 2, h = i & 3;
    float m = fdec(segm[(size_t)snk[e] * NH + h]);
    float a = expf(watt[i] - m);
    watt[i] = a;
    atomicAdd(&segn[(size_t)snk[e] * NH + h], a + 1e-12f);
  }
}

__global__ void k_scatter(const int* __restrict__ snk,
                          const float* __restrict__ watt,
                          const float* __restrict__ segn,
                          const float* __restrict__ nupd, float* upd) {
  long i = (long)blockIdx.x * 256 + threadIdx.x;
  if (i < (long)E_EDGES * DD) {
    int e = (int)(i >> 7), d = (int)(i & 127);
    int s = snk[e];
    float am = 0.0f;
#pragma unroll
    for (int h = 0; h < NH; ++h)
      am += watt[(size_t)e * NH + h] / segn[(size_t)s * NH + h];
    am *= 0.25f;   // mean over heads
    atomicAdd(&upd[(size_t)s * DD + d], am * nupd[i]);
  }
}

// ---------------------------------------------------------------------------
// Node update: bb1=LN(bb+upd); bb2=gelu(bb1@d1+b)@d2+b; bb=LN(bb2+upd)
// One 16-node tile per block; FFN 128->512->128 via WMMA.
// ---------------------------------------------------------------------------
__global__ __launch_bounds__(128) void k_node(
    const float* __restrict__ upd,
    const bf16_t* __restrict__ d1t, const bf16_t* __restrict__ d2t,
    const float* __restrict__ d1b, const float* __restrict__ d2b,
    const float* __restrict__ lng, const float* __restrict__ lnb,
    float* bbf, bf16_t* bbh) {
  __shared__ __attribute__((aligned(16))) bf16_t sA[16 * 136];
  __shared__ __attribute__((aligned(16))) bf16_t sH[16 * 520];
  __shared__ float sF[16 * 132];
  int tid = threadIdx.x, lane = tid & 31, w = tid >> 5;
  int nbm = blockIdx.x * 16;

  // phase 1: LN(bb + upd) -> sA (bf16). wave handles 4 rows, lane 4 cols.
  for (int rr = 0; rr < 4; ++rr) {
    int r = w * 4 + rr, n = nbm + r;
    float t[4], s = 0.0f, s2 = 0.0f;
#pragma unroll
    for (int j = 0; j < 4; ++j) {
      int ci = lane * 4 + j;
      float v = bbf[(size_t)n * DD + ci] + upd[(size_t)n * DD + ci];
      t[j] = v; s += v; s2 += v * v;
    }
    for (int m = 1; m < 32; m <<= 1) { s += __shfl_xor(s, m, 32); s2 += __shfl_xor(s2, m, 32); }
    float mean = s * (1.0f / 128.0f);
    float var  = s2 * (1.0f / 128.0f) - mean * mean;
    float rs   = rsqrtf(var + 1e-5f);
#pragma unroll
    for (int j = 0; j < 4; ++j) {
      int ci = lane * 4 + j;
      sA[r * 136 + ci] = tobf((t[j] - mean) * rs * lng[ci] + lnb[ci]);
    }
  }
  __syncthreads();

  // FFN GEMM1: [16,128]@[128,512], gelu -> sH
  v8f c[8];
#pragma unroll
  for (int t = 0; t < 8; ++t)
#pragma unroll
    for (int i = 0; i < 8; ++i) c[t][i] = 0.0f;
  for (int ks = 0; ks < 4; ++ks) {
    v16bf a = load_a_frag(sA, 136, ks, lane);
#pragma unroll
    for (int t = 0; t < 8; ++t) {
      v16bf b = load_b_frag(d1t, DD, w * 128 + t * 16, ks, lane);
      c[t] = wmma_bf16(a, b, c[t]);
    }
  }
  int rbase = (lane >> 4) * 8;
#pragma unroll
  for (int t = 0; t < 8; ++t) {
    int col = w * 128 + t * 16 + (lane & 15);
    float bias = d1b[col];
#pragma unroll
    for (int v = 0; v < 8; ++v)
      sH[(rbase + v) * 520 + col] = tobf(gelu_f(c[t][v] + bias));
  }
  __syncthreads();

  // FFN GEMM2: [16,512]@[512,128] + bias -> sF (fp32)
  v8f c2[2];
#pragma unroll
  for (int t = 0; t < 2; ++t)
#pragma unroll
    for (int i = 0; i < 8; ++i) c2[t][i] = 0.0f;
  for (int ks = 0; ks < 16; ++ks) {
    v16bf a = load_a_frag(sH, 520, ks, lane);
#pragma unroll
    for (int t = 0; t < 2; ++t) {
      v16bf b = load_b_frag(d2t, 512, w * 32 + t * 16, ks, lane);
      c2[t] = wmma_bf16(a, b, c2[t]);
    }
  }
#pragma unroll
  for (int t = 0; t < 2; ++t) {
    int col = w * 32 + t * 16 + (lane & 15);
    float bias = d2b[col];
#pragma unroll
    for (int v = 0; v < 8; ++v)
      sF[(rbase + v) * 132 + col] = c2[t][v] + bias;
  }
  __syncthreads();

  // phase 4: LN(bb2 + upd) -> bb (fp32 + bf16)
  for (int rr = 0; rr < 4; ++rr) {
    int r = w * 4 + rr, n = nbm + r;
    float t[4], s = 0.0f, s2 = 0.0f;
#pragma unroll
    for (int j = 0; j < 4; ++j) {
      int ci = lane * 4 + j;
      float v = sF[r * 132 + ci] + upd[(size_t)n * DD + ci];
      t[j] = v; s += v; s2 += v * v;
    }
    for (int m = 1; m < 32; m <<= 1) { s += __shfl_xor(s, m, 32); s2 += __shfl_xor(s2, m, 32); }
    float mean = s * (1.0f / 128.0f);
    float var  = s2 * (1.0f / 128.0f) - mean * mean;
    float rs   = rsqrtf(var + 1e-5f);
#pragma unroll
    for (int j = 0; j < 4; ++j) {
      int ci = lane * 4 + j;
      float y = (t[j] - mean) * rs * lng[ci] + lnb[ci];
      bbf[(size_t)n * DD + ci] = y;
      bbh[(size_t)n * DD + ci] = tobf(y);
    }
  }
}

// ---------------------------------------------------------------------------
// Edge update: ef=concat(bb[src],bb[snk],eattr); eupd = MLP(ef);
// eattr = LN(eattr + eupd).
// ---------------------------------------------------------------------------
__global__ __launch_bounds__(128) void k_edge_upd(
    const int* __restrict__ src, const int* __restrict__ snk,
    const bf16_t* __restrict__ bbh,
    const bf16_t* __restrict__ e1t, const bf16_t* __restrict__ e2t,
    const bf16_t* __restrict__ e3t,
    const float* __restrict__ e1b, const float* __restrict__ e2b,
    const float* __restrict__ e3b,
    const float* __restrict__ eng, const float* __restrict__ enb,
    float* eaf, bf16_t* eah) {
  __shared__ __attribute__((aligned(16))) bf16_t sA[16 * 392];
  __shared__ __attribute__((aligned(16))) bf16_t sM[16 * 136];
  __shared__ __attribute__((aligned(16))) bf16_t sM2[16 * 136];
  __shared__ float sF[16 * 132];
  int tid = threadIdx.x, lane = tid & 31, w = tid >> 5;
  int eb = blockIdx.x * 16;

  __builtin_prefetch(e1t + (size_t)w * 32 * DIN_, 0, 1);

  gather_feats(sA, eb, tid, src, snk, bbh, eah);
  __syncthreads();

  int rbase = (lane >> 4) * 8;
  v8f c2[2];

  // e1: [16,384]@[384,128], gelu
#pragma unroll
  for (int t = 0; t < 2; ++t)
#pragma unroll
    for (int i = 0; i < 8; ++i) c2[t][i] = 0.0f;
  for (int ks = 0; ks < 12; ++ks) {
    v16bf a = load_a_frag(sA, 392, ks, lane);
#pragma unroll
    for (int t = 0; t < 2; ++t) {
      v16bf b = load_b_frag(e1t, DIN_, w * 32 + t * 16, ks, lane);
      c2[t] = wmma_bf16(a, b, c2[t]);
    }
  }
#pragma unroll
  for (int t = 0; t < 2; ++t) {
    int col = w * 32 + t * 16 + (lane & 15);
    float bias = e1b[col];
#pragma unroll
    for (int v = 0; v < 8; ++v)
      sM[(rbase + v) * 136 + col] = tobf(gelu_f(c2[t][v] + bias));
  }
  __syncthreads();

  // e2: 128->128, gelu
#pragma unroll
  for (int t = 0; t < 2; ++t)
#pragma unroll
    for (int i = 0; i < 8; ++i) c2[t][i] = 0.0f;
  for (int ks = 0; ks < 4; ++ks) {
    v16bf a = load_a_frag(sM, 136, ks, lane);
#pragma unroll
    for (int t = 0; t < 2; ++t) {
      v16bf b = load_b_frag(e2t, DD, w * 32 + t * 16, ks, lane);
      c2[t] = wmma_bf16(a, b, c2[t]);
    }
  }
#pragma unroll
  for (int t = 0; t < 2; ++t) {
    int col = w * 32 + t * 16 + (lane & 15);
    float bias = e2b[col];
#pragma unroll
    for (int v = 0; v < 8; ++v)
      sM2[(rbase + v) * 136 + col] = tobf(gelu_f(c2[t][v] + bias));
  }
  __syncthreads();

  // e3: 128->128 + bias -> sF
#pragma unroll
  for (int t = 0; t < 2; ++t)
#pragma unroll
    for (int i = 0; i < 8; ++i) c2[t][i] = 0.0f;
  for (int ks = 0; ks < 4; ++ks) {
    v16bf a = load_a_frag(sM2, 136, ks, lane);
#pragma unroll
    for (int t = 0; t < 2; ++t) {
      v16bf b = load_b_frag(e3t, DD, w * 32 + t * 16, ks, lane);
      c2[t] = wmma_bf16(a, b, c2[t]);
    }
  }
#pragma unroll
  for (int t = 0; t < 2; ++t) {
    int col = w * 32 + t * 16 + (lane & 15);
    float bias = e3b[col];
#pragma unroll
    for (int v = 0; v < 8; ++v)
      sF[(rbase + v) * 132 + col] = c2[t][v] + bias;
  }
  __syncthreads();

  // residual + LN -> eattr (fp32 + bf16)
  for (int rr = 0; rr < 4; ++rr) {
    int r = w * 4 + rr, e = eb + r;
    float t[4], s = 0.0f, s2 = 0.0f;
#pragma unroll
    for (int j = 0; j < 4; ++j) {
      int ci = lane * 4 + j;
      float v = eaf[(size_t)e * DD + ci] + sF[r * 132 + ci];
      t[j] = v; s += v; s2 += v * v;
    }
    for (int m = 1; m < 32; m <<= 1) { s += __shfl_xor(s, m, 32); s2 += __shfl_xor(s2, m, 32); }
    float mean = s * (1.0f / 128.0f);
    float var  = s2 * (1.0f / 128.0f) - mean * mean;
    float rs   = rsqrtf(var + 1e-5f);
#pragma unroll
    for (int j = 0; j < 4; ++j) {
      int ci = lane * 4 + j;
      float y = (t[j] - mean) * rs * eng[ci] + enb[ci];
      eaf[(size_t)e * DD + ci] = y;
      eah[(size_t)e * DD + ci] = tobf(y);
    }
  }
}

__global__ void k_out(const float* __restrict__ bbf,
                      const float* __restrict__ eaf, float* out) {
  long i = (long)blockIdx.x * 256 + threadIdx.x;
  long nb = (long)N_NODES * DD;
  long ne = (long)E_EDGES * DD;
  if (i < nb) out[i] = bbf[i];
  else if (i < nb + ne) out[i] = eaf[i - nb];
}

// ---------------------------------------------------------------------------
// Host launch
// ---------------------------------------------------------------------------
extern "C" void kernel_launch(void* const* d_in, const int* in_sizes, int n_in,
                              void* d_out, int out_size, void* d_ws,
                              size_t ws_size, hipStream_t stream) {
  (void)in_sizes; (void)n_in; (void)out_size; (void)ws_size;
  const float* x     = (const float*)d_in[0];
  const int*   ei    = (const int*)d_in[1];
  const float* dist  = (const float*)d_in[2];
  const float* eembw = (const float*)d_in[3];
  const float* eembb = (const float*)d_in[4];
  const float* aWw   = (const float*)d_in[5];
  const float* aWb   = (const float*)d_in[6];
  const float* aAw   = (const float*)d_in[7];
  const float* aAb   = (const float*)d_in[8];
  const float* n1w   = (const float*)d_in[9];
  const float* n1b   = (const float*)d_in[10];
  const float* n2w   = (const float*)d_in[11];
  const float* n2b   = (const float*)d_in[12];
  const float* n3w   = (const float*)d_in[13];
  const float* n3b   = (const float*)d_in[14];
  const float* d1w   = (const float*)d_in[15];
  const float* d1b   = (const float*)d_in[16];
  const float* d2w   = (const float*)d_in[17];
  const float* d2b   = (const float*)d_in[18];
  const float* e1w   = (const float*)d_in[19];
  const float* e1b   = (const float*)d_in[20];
  const float* e2w   = (const float*)d_in[21];
  const float* e2b   = (const float*)d_in[22];
  const float* e3w   = (const float*)d_in[23];
  const float* e3b   = (const float*)d_in[24];
  const float* lng   = (const float*)d_in[25];
  const float* lnb   = (const float*)d_in[26];
  const float* eng   = (const float*)d_in[27];
  const float* enb   = (const float*)d_in[28];

  char* wsp = (char*)d_ws;
  auto alloc = [&](size_t bytes) -> void* {
    void* p = (void*)wsp;
    wsp += (bytes + 255) & ~(size_t)255;
    return p;
  };
  float*    eaf  = (float*)alloc((size_t)E_EDGES * DD * 4);
  bf16_t*   eah  = (bf16_t*)alloc((size_t)E_EDGES * DD * 2);
  float*    nupd = (float*)alloc((size_t)E_EDGES * DD * 4);
  float*    watt = (float*)alloc((size_t)E_EDGES * NH * 4);
  float*    bbf  = (float*)alloc((size_t)N_NODES * DD * 4);
  bf16_t*   bbh  = (bf16_t*)alloc((size_t)N_NODES * DD * 2);
  unsigned* segm = (unsigned*)alloc((size_t)N_NODES * NH * 4);
  float*    segn = (float*)alloc((size_t)N_NODES * NH * 4);
  float*    upd  = (float*)alloc((size_t)N_NODES * DD * 4);
  bf16_t*   W1t  = (bf16_t*)alloc((size_t)640 * DIN_ * 2);
  bf16_t*   n2t  = (bf16_t*)alloc((size_t)DD * DD * 2);
  bf16_t*   n3t  = (bf16_t*)alloc((size_t)DD * DD * 2);
  bf16_t*   d1t  = (bf16_t*)alloc((size_t)512 * DD * 2);
  bf16_t*   d2t  = (bf16_t*)alloc((size_t)DD * 512 * 2);
  bf16_t*   e1t  = (bf16_t*)alloc((size_t)DD * DIN_ * 2);
  bf16_t*   e2t  = (bf16_t*)alloc((size_t)DD * DD * 2);
  bf16_t*   e3t  = (bf16_t*)alloc((size_t)DD * DD * 2);

  const int* srcI = ei;
  const int* snkI = ei + E_EDGES;

  k_copy_x<<<(N_NODES * DD + 255) / 256, 256, 0, stream>>>(x, bbf, bbh);
  k_rbf<<<E_EDGES / 8, 128, 0, stream>>>(dist, eembw, eembb, eaf, eah);

  for (int l = 0; l < NL; ++l) {
    // pre-transpose this layer's weights to bf16 [n][k]
    for (int h = 0; h < NH; ++h)
      kT<<<(DIN_ * DD + 255) / 256, 256, 0, stream>>>(
          W1t + (size_t)h * DD * DIN_, aWw + ((size_t)l * NH + h) * DIN_ * DD,
          DIN_, DD);
    kT<<<(DIN_ * DD + 255) / 256, 256, 0, stream>>>(
        W1t + (size_t)512 * DIN_, n1w + (size_t)l * DIN_ * DD, DIN_, DD);
    kT<<<(DD * DD + 255) / 256, 256, 0, stream>>>(n2t, n2w + (size_t)l * DD * DD, DD, DD);
    kT<<<(DD * DD + 255) / 256, 256, 0, stream>>>(n3t, n3w + (size_t)l * DD * DD, DD, DD);
    kT<<<(DD * 512 + 255) / 256, 256, 0, stream>>>(d1t, d1w + (size_t)l * DD * 512, DD, 512);
    kT<<<(512 * DD + 255) / 256, 256, 0, stream>>>(d2t, d2w + (size_t)l * 512 * DD, 512, DD);
    kT<<<(DIN_ * DD + 255) / 256, 256, 0, stream>>>(e1t, e1w + (size_t)l * DIN_ * DD, DIN_, DD);
    kT<<<(DD * DD + 255) / 256, 256, 0, stream>>>(e2t, e2w + (size_t)l * DD * DD, DD, DD);
    kT<<<(DD * DD + 255) / 256, 256, 0, stream>>>(e3t, e3w + (size_t)l * DD * DD, DD, DD);

    k_init<<<(N_NODES * DD + 255) / 256, 256, 0, stream>>>(segm, segn, upd);

    k_edge_stage1<<<E_EDGES / 16, 128, 0, stream>>>(
        srcI, snkI, bbh, eah, W1t, n2t, n3t,
        aWb + (size_t)l * NH * DD, aAw + (size_t)l * NH * DD, aAb + (size_t)l * NH,
        n1b + (size_t)l * DD, n2b + (size_t)l * DD, n3b + (size_t)l * DD,
        watt, nupd);

    k_seg_max<<<(E_EDGES * NH + 255) / 256, 256, 0, stream>>>(snkI, watt, segm);
    k_att_exp<<<(E_EDGES * NH + 255) / 256, 256, 0, stream>>>(snkI, watt, segm, segn);
    k_scatter<<<(int)(((long)E_EDGES * DD + 255) / 256), 256, 0, stream>>>(
        snkI, watt, segn, nupd, upd);

    k_node<<<N_NODES / 16, 128, 0, stream>>>(
        upd, d1t, d2t, d1b + (size_t)l * 512, d2b + (size_t)l * DD,
        lng + (size_t)l * DD, lnb + (size_t)l * DD, bbf, bbh);

    k_edge_upd<<<E_EDGES / 16, 128, 0, stream>>>(
        srcI, snkI, bbh, e1t, e2t, e3t,
        e1b + (size_t)l * DD, e2b + (size_t)l * DD, e3b + (size_t)l * DD,
        eng + (size_t)l * DD, enb + (size_t)l * DD, eaf, eah);
  }

  long tot = (long)N_NODES * DD + (long)E_EDGES * DD;
  k_out<<<(int)((tot + 255) / 256), 256, 0, stream>>>(bbf, eaf, (float*)d_out);
}
